// CrossAttention_71906342470083
// MI455X (gfx1250) — matrix operand order repaired and checked by reading, not compiled, not run on hardware
//
#include <hip/hip_runtime.h>
#include <hip/hip_bf16.h>
#include <math.h>

typedef __bf16 bf16;
typedef __attribute__((ext_vector_type(16))) __bf16 v16bf;
typedef __attribute__((ext_vector_type(8)))  __bf16 v8bf;
typedef __attribute__((ext_vector_type(4)))  __bf16 v4bf;
typedef __attribute__((ext_vector_type(8)))  float  v8f;
typedef __attribute__((ext_vector_type(4)))  float  v4f;
typedef __attribute__((ext_vector_type(4)))  unsigned int v4u;
typedef __attribute__((ext_vector_type(8)))  int    v8i;
typedef __attribute__((ext_vector_type(4)))  int    v4i;

#define BATCH 32
#define SEQQ  512
#define SEQK  512
#define DM    512
#define NH    8
#define HD    64
#define ATT_SCALE 0.125f   // 1/sqrt(64)

// ---------------------------------------------------------------------------
// WMMA fragment loaders (wave32, 16x16x32 bf16), CDNA5 ISA 7.12.2 layouts.
// Both compile to 2x ds_load_b128 per fragment.
// ---------------------------------------------------------------------------
__device__ __forceinline__ v16bf frag_a_rowmajor(const bf16* p, int ld, int lane) {
  int m  = lane & 15;
  int kb = (lane >> 4) ? 8 : 0;
  const bf16* row = p + m * ld + kb;
  union { v16bf v; v8bf h[2]; } u;
  u.h[0] = *(const v8bf*)(row);        // K = kb .. kb+7
  u.h[1] = *(const v8bf*)(row + 16);   // K = 16+kb .. 16+kb+7
  return u.v;
}

__device__ __forceinline__ v16bf frag_b_nmajor(const bf16* p, int ld, int lane) {
  int n  = lane & 15;
  int kb = (lane >> 4) ? 8 : 0;
  const bf16* col = p + n * ld + kb;
  union { v16bf v; v8bf h[2]; } u;
  u.h[0] = *(const v8bf*)(col);
  u.h[1] = *(const v8bf*)(col + 16);
  return u.v;
}

__device__ __forceinline__ v8f wmma_bf16(v16bf a, v16bf b, v8f c) {
  return __builtin_amdgcn_wmma_f32_16x16x32_bf16(false, a, false, b, (short)0, c,
                                                 false, false);
}

// ---------------------------------------------------------------------------
// Tensor Data Mover: 2D bf16 tile (tile_d0 elems/row x tile_d1 rows) from
// global to LDS, with LDS row padding 128B->144B (pad_interval=32 DWORDs,
// pad_amount=4 DWORDs). D# layout per CDNA5 ISA ch.8 (groups 0..3).
// Issue from a single wave; completion via s_wait_tensorcnt.
// ---------------------------------------------------------------------------
__device__ __forceinline__ unsigned lds_addr_of(const void* p) {
  // low 32 bits of a flat shared-aperture address are the LDS byte address
  return (unsigned)(uintptr_t)p;
}

__device__ __forceinline__ void tdm_load_2d_bf16(
    unsigned lds_byte_addr, const bf16* gptr, unsigned tile_d0,
    unsigned tile_d1, unsigned tensor_d0, unsigned tensor_d1,
    unsigned stride_d0) {
  unsigned long long ga = (unsigned long long)(uintptr_t)gptr;
  v4u g0;
  g0[0] = 1u;                                        // count=1, user mode
  g0[1] = lds_byte_addr;                             // lds_addr
  g0[2] = (unsigned)(ga & 0xFFFFFFFFu);              // global_addr[31:0]
  g0[3] = (unsigned)((ga >> 32) & 0x01FFFFFFu)       // global_addr[56:32]
        | (2u << 30);                                // type=2 ("image")
  v8i g1;
  g1[0] = (int)((1u << 16)                           // data_size = 2 bytes
              | (1u << 20)                           // pad_enable
              | (4u << 22)                           // pad_interval: 32 DWORDs
              | (3u << 25));                         // pad_amount: 4 DWORDs
  g1[1] = (int)((tensor_d0 & 0xFFFFu) << 16);        // tensor_dim0 bits 79:48
  g1[2] = (int)(((tensor_d0 >> 16) & 0xFFFFu)
              | ((tensor_d1 & 0xFFFFu) << 16));      // tensor_dim1 bits 111:80
  g1[3] = (int)(((tensor_d1 >> 16) & 0xFFFFu)
              | ((tile_d0 & 0xFFFFu) << 16));        // tile_dim0 bits 127:112
  g1[4] = (int)(tile_d1 & 0xFFFFu);                  // tile_dim1; tile_dim2=0
  g1[5] = (int)stride_d0;                            // tensor_dim0_stride lo32
  g1[6] = 0;
  g1[7] = 0;
  v4i g2 = (v4i){0, 0, 0, 0};
  v4i g3 = (v4i){0, 0, 0, 0};
  v8i g4 = (v8i){0, 0, 0, 0, 0, 0, 0, 0};
  __builtin_amdgcn_tensor_load_to_lds(g0, g1, g2, g3, g4, 0);
}

// ---------------------------------------------------------------------------
// Kernel 0: f32 -> bf16 conversion (weights)
// ---------------------------------------------------------------------------
__global__ void cvt_bf16_kernel(const float* __restrict__ src,
                                bf16* __restrict__ dst, int n) {
  int i = blockIdx.x * blockDim.x + threadIdx.x;
  if (i < n) dst[i] = (bf16)src[i];
}

// ---------------------------------------------------------------------------
// Kernel 1: projection GEMM  Out[b,h,s,d] = (X[16384,512] @ W[512,512])
// ---------------------------------------------------------------------------
__global__ __launch_bounds__(256) void proj_gemm_kernel(
    const float* __restrict__ X, const bf16* __restrict__ W,
    bf16* __restrict__ Out) {
  __shared__ bf16 As[128 * 40];   // [m][k]
  __shared__ bf16 Bs[128 * 40];   // [n][k]  (transposed at stage-in)

  int tid = threadIdx.x;
  int wave = tid >> 5, lane = tid & 31;
  int m0 = blockIdx.y * 128;
  int n0 = blockIdx.x * 128;
  int wm = (wave & 3) * 32;
  int wn = (wave >> 2) * 64;

  v8f acc[2][4];
#pragma unroll
  for (int i = 0; i < 2; ++i)
#pragma unroll
    for (int j = 0; j < 4; ++j) acc[i][j] = (v8f){};

  for (int kb = 0; kb < DM; kb += 32) {
    __syncthreads();
#pragma unroll
    for (int i = 0; i < 4; ++i) {        // A tile 128x32: float4, cvt, b64 store
      int e = i * 1024 + tid * 4;
      int r = e >> 5, k = e & 31;
      v4f f = *(const v4f*)&X[(size_t)(m0 + r) * DM + kb + k];
      v4bf o;
#pragma unroll
      for (int j = 0; j < 4; ++j) o[j] = (bf16)f[j];
      *(v4bf*)&As[r * 40 + k] = o;
    }
#pragma unroll
    for (int i = 0; i < 2; ++i) {        // W tile transposed into [n][k]
      int e = i * 2048 + tid * 8;
      int k = e >> 7, n = e & 127;
      v8bf w = *(const v8bf*)&W[(size_t)(kb + k) * DM + n0 + n];
#pragma unroll
      for (int j = 0; j < 8; ++j) Bs[(n + j) * 40 + k] = w[j];
    }
    if (kb + 32 < DM) {
      __builtin_prefetch(&X[(size_t)(m0 + (tid >> 3)) * DM + kb + 32], 0, 1);
      __builtin_prefetch(&W[(size_t)(kb + 32 + (tid >> 5)) * DM + n0 + ((tid & 31) << 2)], 0, 1);
    }
    __syncthreads();

    v16bf afr[2], bfr[4];
#pragma unroll
    for (int mi = 0; mi < 2; ++mi)
      afr[mi] = frag_a_rowmajor(As + (wm + mi * 16) * 40, 40, lane);
#pragma unroll
    for (int ni = 0; ni < 4; ++ni)
      bfr[ni] = frag_b_nmajor(Bs + (wn + ni * 16) * 40, 40, lane);
#pragma unroll
    for (int mi = 0; mi < 2; ++mi)
#pragma unroll
      for (int ni = 0; ni < 4; ++ni)
        acc[mi][ni] = wmma_bf16(afr[mi], bfr[ni], acc[mi][ni]);
  }

#pragma unroll
  for (int mi = 0; mi < 2; ++mi)
#pragma unroll
    for (int ni = 0; ni < 4; ++ni) {
      int rowb = m0 + wm + mi * 16 + ((lane >> 4) ? 8 : 0);
      int col  = n0 + wn + ni * 16 + (lane & 15);
      int h = col >> 6, d = col & 63;
#pragma unroll
      for (int r = 0; r < 8; ++r) {
        int row = rowb + r;
        int b = row >> 9, s = row & 511;
        Out[(((size_t)b * NH + h) * SEQQ + s) * HD + d] = (bf16)acc[mi][ni][r];
      }
    }
}

// ---------------------------------------------------------------------------
// Kernel 2: fused attention for one (b, h, 64-row q tile).
// Q tile + K chunks staged by the Tensor Data Mover (double-buffered K).
// ---------------------------------------------------------------------------
__global__ __launch_bounds__(256) void attn_kernel(
    const bf16* __restrict__ Qb, const bf16* __restrict__ Kb,
    const bf16* __restrict__ Vb, const float* __restrict__ rel,
    const int* __restrict__ mask, bf16* __restrict__ ctx) {
  __shared__ bf16  Qs[64 * 72];        // [q][d], TDM-padded rows
  __shared__ bf16  Kdb[2][128 * 72];   // double-buffered K chunks [sk][d]
  __shared__ bf16  Vs[64 * 136];       // V chunk transposed [d][sk]
  __shared__ float Ss[64 * 516];
  __shared__ bf16  Ps[64 * 520];
  __shared__ float red[64 * 4];

  int tid = threadIdx.x;
  int wave = tid >> 5, lane = tid & 31;
  int q0 = blockIdx.x * 64;
  int h  = blockIdx.y;
  int b  = blockIdx.z;

  const bf16* Qp = Qb + ((size_t)b * NH + h) * SEQQ * HD;
  const bf16* Kp = Kb + ((size_t)b * NH + h) * SEQK * HD;
  const bf16* Vp = Vb + ((size_t)b * NH + h) * SEQK * HD;

  // TDM: Q tile (64 rows) + first K chunk (128 rows), issued by wave 0 only
  if (wave == 0) {
    tdm_load_2d_bf16(lds_addr_of(Qs), Qp + (size_t)q0 * HD,
                     HD, 64, HD, SEQQ, HD);
    tdm_load_2d_bf16(lds_addr_of(Kdb[0]), Kp, HD, 128, HD, SEQK, HD);
    __builtin_amdgcn_s_wait_tensorcnt(0);
  }
  __syncthreads();

  int wm = (wave & 3) * 16;              // q sub-strip of this wave
  int wn = (wave >> 2) * 64;             // sk column group (phase 1)

  v16bf a0 = frag_a_rowmajor(Qs + wm * 72, 72, lane);
  v16bf a1 = frag_a_rowmajor(Qs + wm * 72 + 32, 72, lane);

  // -------- phase 1: S = QK^T * scale + bias, mask -------------------------
  for (int c = 0; c < 4; ++c) {
    int sk0 = c * 128;
    // async prefetch of next K chunk into the other buffer (TDM)
    if (wave == 0 && c < 3)
      tdm_load_2d_bf16(lds_addr_of(Kdb[(c + 1) & 1]),
                       Kp + (size_t)(sk0 + 128) * HD, HD, 128, HD, SEQK, HD);
    const bf16* Kc = Kdb[c & 1];

#pragma unroll
    for (int ni = 0; ni < 4; ++ni) {
      v16bf b0 = frag_b_nmajor(Kc + (wn + ni * 16) * 72, 72, lane);
      v16bf b1 = frag_b_nmajor(Kc + (wn + ni * 16) * 72 + 32, 72, lane);
      v8f c8 = (v8f){};
      c8 = wmma_bf16(a0, b0, c8);
      c8 = wmma_bf16(a1, b1, c8);

      int srowb = wm + ((lane >> 4) ? 8 : 0);
      int kc    = sk0 + wn + ni * 16 + (lane & 15);
#pragma unroll
      for (int r = 0; r < 8; ++r) {
        int qr = q0 + srowb + r;
        float sv = c8[r] * ATT_SCALE + rel[(size_t)qr * SEQK + kc];
        int mv = mask[((size_t)b * SEQQ + qr) * SEQK + kc];
        if (mv == 0) sv = -1e30f;
        Ss[(srowb + r) * 516 + kc] = sv;
      }
    }
    if (wave == 0 && c < 3) __builtin_amdgcn_s_wait_tensorcnt(0);
    __syncthreads();
  }

  // -------- softmax over 512 columns (4 threads per row) ------------------
  {
    int row = tid >> 2, seg = tid & 3;
    float* sp = Ss + row * 516 + seg * 128;
    float mx = -1e30f;
    for (int j = 0; j < 128; ++j) mx = fmaxf(mx, sp[j]);
    red[row * 4 + seg] = mx;
    __syncthreads();
    float rm = fmaxf(fmaxf(red[row * 4], red[row * 4 + 1]),
                     fmaxf(red[row * 4 + 2], red[row * 4 + 3]));
    __syncthreads();
    float sum = 0.f;
    for (int j = 0; j < 128; ++j) {
      float e = __expf(sp[j] - rm);
      sp[j] = e;
      sum += e;
    }
    red[row * 4 + seg] = sum;
    __syncthreads();
    float rs = red[row * 4] + red[row * 4 + 1] + red[row * 4 + 2] +
               red[row * 4 + 3];
    float inv = 1.f / rs;
    bf16* pp = Ps + row * 520 + seg * 128;
    for (int j = 0; j < 128; ++j) pp[j] = (bf16)(sp[j] * inv);
  }

  // -------- phase 2: ctx = P @ V ------------------------------------------
  int wn2 = (wave >> 2) * 32;
  v8f o0 = (v8f){}, o1 = (v8f){};
  for (int c = 0; c < 4; ++c) {
    int sk0 = c * 128;
    __syncthreads();
    // V chunk 128x64 stored transposed [d][sk] (ld=136) for n-major frags
#pragma unroll
    for (int i = 0; i < 4; ++i) {
      int e = i * 2048 + tid * 8;
      int r = e >> 6, d = e & 63;
      v8bf v = *(const v8bf*)&Vp[(size_t)(sk0 + r) * HD + d];
#pragma unroll
      for (int j = 0; j < 8; ++j) Vs[(d + j) * 136 + r] = v[j];
    }
    if (c < 3)
      __builtin_prefetch(&Vp[(size_t)(sk0 + 128 + (tid >> 1)) * HD], 0, 1);
    __syncthreads();
#pragma unroll
    for (int ks = 0; ks < 4; ++ks) {
      v16bf a  = frag_a_rowmajor(Ps + wm * 520 + sk0 + ks * 32, 520, lane);
      v16bf b0 = frag_b_nmajor(Vs + wn2 * 136 + ks * 32, 136, lane);
      v16bf b1 = frag_b_nmajor(Vs + (wn2 + 16) * 136 + ks * 32, 136, lane);
      o0 = wmma_bf16(a, b0, o0);
      o1 = wmma_bf16(a, b1, o1);
    }
  }

  // store ctx in [B, S, H*HD] bf16 for the output GEMM
  int qrb = q0 + wm + ((lane >> 4) ? 8 : 0);
#pragma unroll
  for (int r = 0; r < 8; ++r) {
    int qr = qrb + r;
    int d0 = wn2 + (lane & 15);
    ctx[((size_t)b * SEQQ + qr) * DM + h * HD + d0]      = (bf16)o0[r];
    ctx[((size_t)b * SEQQ + qr) * DM + h * HD + d0 + 16] = (bf16)o1[r];
  }
}

// ---------------------------------------------------------------------------
// Kernel 3: output GEMM  out[16384,512] = ctx_bf16 @ Wo_bf16 + bo  (f32 out)
// ---------------------------------------------------------------------------
__global__ __launch_bounds__(256) void out_gemm_kernel(
    const bf16* __restrict__ X, const bf16* __restrict__ W,
    const float* __restrict__ bias, float* __restrict__ Out) {
  __shared__ bf16 As[128 * 40];   // [m][k]
  __shared__ bf16 Bs[128 * 40];   // [n][k]

  int tid = threadIdx.x;
  int wave = tid >> 5, lane = tid & 31;
  int m0 = blockIdx.y * 128;
  int n0 = blockIdx.x * 128;
  int wm = (wave & 3) * 32;
  int wn = (wave >> 2) * 64;

  v8f acc[2][4];
#pragma unroll
  for (int i = 0; i < 2; ++i)
#pragma unroll
    for (int j = 0; j < 4; ++j) acc[i][j] = (v8f){};

  for (int kb = 0; kb < DM; kb += 32) {
    __syncthreads();
#pragma unroll
    for (int i = 0; i < 2; ++i) {        // A tile 128x32 bf16, b128 copies
      int e = i * 2048 + tid * 8;
      int r = e >> 5, k = e & 31;
      *(v8bf*)&As[r * 40 + k] = *(const v8bf*)&X[(size_t)(m0 + r) * DM + kb + k];
    }
#pragma unroll
    for (int i = 0; i < 2; ++i) {        // W tile transposed into [n][k]
      int e = i * 2048 + tid * 8;
      int k = e >> 7, n = e & 127;
      v8bf w = *(const v8bf*)&W[(size_t)(kb + k) * DM + n0 + n];
#pragma unroll
      for (int j = 0; j < 8; ++j) Bs[(n + j) * 40 + k] = w[j];
    }
    if (kb + 32 < DM) {
      __builtin_prefetch(&X[(size_t)(m0 + (tid >> 1)) * DM + kb + 32], 0, 1);
      __builtin_prefetch(&W[(size_t)(kb + 32 + (tid >> 5)) * DM + n0 + ((tid & 31) << 2)], 0, 1);
    }
    __syncthreads();

    v16bf afr[2], bfr[4];
#pragma unroll
    for (int mi = 0; mi < 2; ++mi)
      afr[mi] = frag_a_rowmajor(As + (wm + mi * 16) * 40, 40, lane);
#pragma unroll
    for (int ni = 0; ni < 4; ++ni)
      bfr[ni] = frag_b_nmajor(Bs + (wn + ni * 16) * 40, 40, lane);
#pragma unroll
    for (int mi = 0; mi < 2; ++mi)
#pragma unroll
      for (int ni = 0; ni < 4; ++ni)
        acc[mi][ni] = wmma_bf16(afr[mi], bfr[ni], acc[mi][ni]);
  }

#pragma unroll
  for (int mi = 0; mi < 2; ++mi)
#pragma unroll
    for (int ni = 0; ni < 4; ++ni) {
      int rowb = m0 + wm + mi * 16 + ((lane >> 4) ? 8 : 0);
      int col  = n0 + wn + ni * 16 + (lane & 15);
      float bv = bias[col];
#pragma unroll
      for (int r = 0; r < 8; ++r)
        Out[(size_t)(rowb + r) * DM + col] = acc[mi][ni][r] + bv;
    }
}

// ---------------------------------------------------------------------------
extern "C" void kernel_launch(void* const* d_in, const int* in_sizes, int n_in,
                              void* d_out, int out_size, void* d_ws,
                              size_t ws_size, hipStream_t stream) {
  (void)in_sizes; (void)n_in; (void)out_size; (void)ws_size;

  const float* query = (const float*)d_in[0];
  const float* key   = (const float*)d_in[1];
  const float* value = (const float*)d_in[2];
  const int*   mask  = (const int*)d_in[3];
  const float* Wq    = (const float*)d_in[4];
  const float* Wk    = (const float*)d_in[5];
  const float* Wv    = (const float*)d_in[6];
  const float* Wo    = (const float*)d_in[7];
  const float* bo    = (const float*)d_in[8];
  const float* rel   = (const float*)d_in[9];
  float* out = (float*)d_out;

  char* ws = (char*)d_ws;
  const size_t qkv_bytes = (size_t)BATCH * NH * SEQQ * HD * sizeof(bf16); // 16 MB
  const size_t w_bytes   = (size_t)DM * DM * sizeof(bf16);                // 512 KB
  bf16* Qb  = (bf16*)ws;                 ws += qkv_bytes;
  bf16* Kb  = (bf16*)ws;                 ws += qkv_bytes;
  bf16* Vb  = (bf16*)ws;                 ws += qkv_bytes;
  bf16* ctx = (bf16*)ws;                 ws += qkv_bytes;
  bf16* Wqb = (bf16*)ws;                 ws += w_bytes;
  bf16* Wkb = (bf16*)ws;                 ws += w_bytes;
  bf16* Wvb = (bf16*)ws;                 ws += w_bytes;
  bf16* Wob = (bf16*)ws;                 ws += w_bytes;

  const int wn = DM * DM;
  cvt_bf16_kernel<<<(wn + 255) / 256, 256, 0, stream>>>(Wq, Wqb, wn);
  cvt_bf16_kernel<<<(wn + 255) / 256, 256, 0, stream>>>(Wk, Wkb, wn);
  cvt_bf16_kernel<<<(wn + 255) / 256, 256, 0, stream>>>(Wv, Wvb, wn);
  cvt_bf16_kernel<<<(wn + 255) / 256, 256, 0, stream>>>(Wo, Wob, wn);

  dim3 gg(4, 128);   // N tiles x M tiles (16384 x 512 GEMM, 128x128 blocks)
  proj_gemm_kernel<<<gg, 256, 0, stream>>>(query, Wqb, Qb);
  proj_gemm_kernel<<<gg, 256, 0, stream>>>(key,   Wkb, Kb);
  proj_gemm_kernel<<<gg, 256, 0, stream>>>(value, Wvb, Vb);

  attn_kernel<<<dim3(SEQQ / 64, NH, BATCH), 256, 0, stream>>>(Qb, Kb, Vb, rel,
                                                              mask, ctx);

  out_gemm_kernel<<<gg, 256, 0, stream>>>(ctx, Wob, bo, out);
}
